// SurfaceTransformerV2_27058293965201
// MI455X (gfx1250) — matrix operand order repaired and checked by reading, not compiled, not run on hardware
//
#include <hip/hip_runtime.h>
#include <stdint.h>

// ---------------- problem constants (match reference) ----------------
#define BATCH   8
#define NPTS    4096
#define DIM     128
#define NPATCH  256
#define KNN     64
#define DFF     1024
#define BP      (BATCH * NPATCH)        // 2048
#define RFINE   (BP * KNN)              // 131072
#define FCK     160                     // featc K padded 133 -> 160
#define W3NP    64                      // w3 N padded 48 -> 64

// ---------------- types for WMMA ----------------
typedef __attribute__((ext_vector_type(16))) __bf16 v16bf;
typedef __attribute__((ext_vector_type(8)))  float  v8f;

union FragBF { uint4 u[2]; v16bf v; };

__device__ __forceinline__ unsigned short f2bf(float f) {
    unsigned int u = __float_as_uint(f);
    unsigned int r = u + 0x7FFFu + ((u >> 16) & 1u);   // RNE
    return (unsigned short)(r >> 16);
}
__device__ __forceinline__ float bf2f(unsigned short h) {
    return __uint_as_float(((unsigned int)h) << 16);
}

// =====================================================================
// 1) FPS: one block per batch. mind[] lives in LDS; 255 argmax rounds.
// =====================================================================
__global__ __launch_bounds__(256)
void fps_kernel(const float* __restrict__ xyz, int* __restrict__ fidx) {
    const int b = blockIdx.x;
    const float* X = xyz + (size_t)b * NPTS * 3;
    __shared__ float mind[NPTS];
    __shared__ float rv[256];
    __shared__ int   ri[256];
    const int t = threadIdx.x;
    for (int n = t; n < NPTS; n += 256) mind[n] = 1e10f;
    if (t == 0) fidx[b * NPATCH] = 0;
    int last = 0;
    __syncthreads();
    for (int s = 1; s < NPATCH; ++s) {
        const float lx = X[last * 3 + 0], ly = X[last * 3 + 1], lz = X[last * 3 + 2];
        float bv = -1.0f; int bi = NPTS;
        for (int n = t; n < NPTS; n += 256) {
            float dx = X[n * 3 + 0] - lx, dy = X[n * 3 + 1] - ly, dz = X[n * 3 + 2] - lz;
            float d = dx * dx + dy * dy + dz * dz;
            float m = fminf(mind[n], d);
            mind[n] = m;
            if (m > bv) { bv = m; bi = n; }
        }
        rv[t] = bv; ri[t] = bi;
        __syncthreads();
        for (int off = 128; off > 0; off >>= 1) {
            if (t < off) {
                if (rv[t + off] > rv[t] || (rv[t + off] == rv[t] && ri[t + off] < ri[t])) {
                    rv[t] = rv[t + off]; ri[t] = ri[t + off];
                }
            }
            __syncthreads();
        }
        last = ri[0];
        if (t == 0) fidx[b * NPATCH + s] = last;
        __syncthreads();
    }
}

// =====================================================================
// 2) KNN top-64: one block per (b,p). d[4096] in LDS, 64 argmin rounds.
//    Also emits grouped_xyz_norm (gathered xyz minus patch center).
// =====================================================================
__global__ __launch_bounds__(256)
void topk_kernel(const float* __restrict__ xyz, const int* __restrict__ fidx,
                 int* __restrict__ gidx, float* __restrict__ gxyz) {
    const int bp = blockIdx.x;
    const int b = bp / NPATCH;
    const float* X = xyz + (size_t)b * NPTS * 3;
    const int ci = fidx[bp];
    const float cx = X[ci * 3 + 0], cy = X[ci * 3 + 1], cz = X[ci * 3 + 2];
    __shared__ float d[NPTS];
    __shared__ float rv[256];
    __shared__ int   ri[256];
    const int t = threadIdx.x;
    for (int n = t; n < NPTS; n += 256) {
        float dx = X[n * 3 + 0] - cx, dy = X[n * 3 + 1] - cy, dz = X[n * 3 + 2] - cz;
        d[n] = dx * dx + dy * dy + dz * dz;
    }
    __syncthreads();
    for (int k = 0; k < KNN; ++k) {
        float bv = 1e31f; int bi = NPTS;
        for (int n = t; n < NPTS; n += 256) {
            float v = d[n];
            if (v < bv) { bv = v; bi = n; }
        }
        rv[t] = bv; ri[t] = bi;
        __syncthreads();
        for (int off = 128; off > 0; off >>= 1) {
            if (t < off) {
                if (rv[t + off] < rv[t] || (rv[t + off] == rv[t] && ri[t + off] < ri[t])) {
                    rv[t] = rv[t + off]; ri[t] = ri[t + off];
                }
            }
            __syncthreads();
        }
        const int sel = ri[0];
        if (t == 0) {
            gidx[(size_t)bp * KNN + k] = sel;
            d[sel] = 1e30f;
            gxyz[((size_t)bp * KNN + k) * 3 + 0] = X[sel * 3 + 0] - cx;
            gxyz[((size_t)bp * KNN + k) * 3 + 1] = X[sel * 3 + 1] - cy;
            gxyz[((size_t)bp * KNN + k) * 3 + 2] = X[sel * 3 + 2] - cz;
        }
        __syncthreads();
    }
}

// =====================================================================
// 3) Gather features + max-pool over K -> patch_feat (stored bf16 for GEMM A)
// =====================================================================
__global__ __launch_bounds__(128)
void gather_max_kernel(const float* __restrict__ feat, const int* __restrict__ gidx,
                       unsigned short* __restrict__ pf) {
    const int bp = blockIdx.x;
    const int b = bp / NPATCH;
    const int c = threadIdx.x;            // 0..127
    const int* gi = gidx + (size_t)bp * KNN;
    float m = -1e30f;
    for (int k = 0; k < KNN; ++k) {
        float f = feat[((size_t)b * NPTS + gi[k]) * DIM + c];
        m = fmaxf(m, f);
    }
    pf[(size_t)bp * DIM + c] = f2bf(m);
}

// =====================================================================
// 4) WMMA bf16 GEMM: C = relu?( (A@Bt^T) * scale[n] + shift[n] )
//    A  [M,K] bf16 row-major.
//    Bt [N,K] bf16 row-major (weights PRE-TRANSPOSED on device) -> all
//    staging is vectorized b128 copies; no scalar ds stores in hot loop.
//    Tile 128x64, BK=32. 8 waves; each wave owns 2x2 16x16 f32 accums.
// =====================================================================
#define GBM 128
#define GBN 64
#define GBK 32
#define LSA 40   // LDS stride (bf16 elems) for A tile  (80B: 16B-aligned, conflict-free)
#define LSB 40   // LDS stride for B tile [n][k]

__global__ __launch_bounds__(256)
void gemm_bf16_wmma(const unsigned short* __restrict__ A,
                    const unsigned short* __restrict__ Bt,
                    const float* __restrict__ scale,
                    const float* __restrict__ shift,
                    float* __restrict__ outF,
                    unsigned short* __restrict__ outB,
                    int M, int N, int K, int relu) {
    __shared__ unsigned short sA[GBM * LSA];
    __shared__ unsigned short sB[GBN * LSB];   // [n][k]

    const int t    = threadIdx.x;
    const int lane = t & 31;
    const int wave = t >> 5;
    const int wm   = wave & 3;    // 4 waves down  -> 32-row strips
    const int wn   = wave >> 2;   // 2 waves across-> 32-col strips
    const int hi   = lane >> 4;   // lane half (0: lanes 0-15, 1: lanes 16-31)
    const int ln   = lane & 15;

    const int m_blk = blockIdx.y * GBM;
    const int n_blk = blockIdx.x * GBN;

    v8f zero = {0.f, 0.f, 0.f, 0.f, 0.f, 0.f, 0.f, 0.f};
    v8f acc[2][2];
    acc[0][0] = zero; acc[0][1] = zero; acc[1][0] = zero; acc[1][1] = zero;

    // staging indices (constant per thread)
    const int ar0 = t >> 2;            // A rows handled: ar0 and ar0+64
    const int ac  = (t & 3) * 8;       // A k-chunk (8 bf16)
    const int bn  = t >> 2;            // B row (n) 0..63
    const int bc  = (t & 3) * 8;       // B k-chunk

    for (int k0 = 0; k0 < K; k0 += GBK) {
        // --- stage A tile 128x32: 2 x uint4 per thread ---
        *(uint4*)&sA[ar0 * LSA + ac] =
            *(const uint4*)&A[(size_t)(m_blk + ar0) * K + k0 + ac];
        *(uint4*)&sA[(ar0 + 64) * LSA + ac] =
            *(const uint4*)&A[(size_t)(m_blk + ar0 + 64) * K + k0 + ac];
        // --- stage B tile 64x32 from pre-transposed Bt: 1 x uint4 per thread ---
        *(uint4*)&sB[bn * LSB + bc] =
            *(const uint4*)&Bt[(size_t)(n_blk + bn) * K + k0 + bc];
        __syncthreads();

        // --- fragments per documented 16-bit layouts ---
        // A 16x32: lane row = lane&15; K chunks [8*hi, 8*hi+7] -> elems 0..7,
        //                              [16+8*hi, 23+8*hi]      -> elems 8..15
        FragBF fa[2], fb[2];
#pragma unroll
        for (int sm = 0; sm < 2; ++sm) {
            int row = wm * 32 + sm * 16 + ln;
            fa[sm].u[0] = *(const uint4*)&sA[row * LSA + 8 * hi];
            fa[sm].u[1] = *(const uint4*)&sA[row * LSA + 16 + 8 * hi];
        }
        // B 32x16: lane col = lane&15; K chunks [8*hi, 8*hi+7] -> elems 0..7, [16+8*hi, 23+8*hi] -> elems 8..15
#pragma unroll
        for (int sn = 0; sn < 2; ++sn) {
            int col = wn * 32 + sn * 16 + ln;
            fb[sn].u[0] = *(const uint4*)&sB[col * LSB + 8 * hi];
            fb[sn].u[1] = *(const uint4*)&sB[col * LSB + 16 + 8 * hi];
        }
#pragma unroll
        for (int sm = 0; sm < 2; ++sm)
#pragma unroll
            for (int sn = 0; sn < 2; ++sn)
                acc[sm][sn] = __builtin_amdgcn_wmma_f32_16x16x32_bf16(
                    false, fa[sm].v, false, fb[sn].v,
                    (short)0, acc[sm][sn], false, false);
        __syncthreads();
    }

    // --- epilogue: C VGPR r -> (M = r + 8*hi, N = lane&15) ---
    const bool toBf = (outB != nullptr);
#pragma unroll
    for (int sm = 0; sm < 2; ++sm) {
#pragma unroll
        for (int sn = 0; sn < 2; ++sn) {
            const int gn = n_blk + wn * 32 + sn * 16 + ln;
            const int rb = m_blk + wm * 32 + sm * 16 + hi * 8;
            const float sc = scale[gn], sh = shift[gn];
#pragma unroll
            for (int r = 0; r < 8; ++r) {
                float v = acc[sm][sn][r] * sc + sh;
                if (relu) v = fmaxf(v, 0.0f);
                size_t o = (size_t)(rb + r) * N + gn;
                if (toBf) outB[o] = f2bf(v);
                else      outF[o] = v;
            }
        }
    }
}

// =====================================================================
// 5) Weight prep: transpose+convert f32 [Ks,Ns] -> bf16 [No,Ko] (zero-pad),
//    and bias / BN folding into per-column (scale, shift).
// =====================================================================
__global__ void cvt_transpose_kernel(const float* __restrict__ src, unsigned short* __restrict__ dst,
                                     int Ko, int No, int Ks, int Ns) {
    // dst[n*Ko + k] = src[k*Ns + n] for n<Ns, k<Ks; else 0
    int i = blockIdx.x * 256 + threadIdx.x;
    if (i >= No * Ko) return;
    int n = i / Ko, k = i % Ko;
    dst[i] = (n < Ns && k < Ks) ? f2bf(src[(size_t)k * Ns + n]) : (unsigned short)0;
}
__global__ void fold_bias_kernel(const float* __restrict__ b, float* __restrict__ sc,
                                 float* __restrict__ sh, int n, int nvalid) {
    int i = blockIdx.x * 256 + threadIdx.x;
    if (i < n) { sc[i] = 1.0f; sh[i] = (i < nvalid) ? b[i] : 0.0f; }
}
__global__ void fold_bn_kernel(const float* __restrict__ cb, const float* __restrict__ g,
                               const float* __restrict__ be, const float* __restrict__ m,
                               const float* __restrict__ v, float* __restrict__ sc,
                               float* __restrict__ sh, int n) {
    int i = blockIdx.x * 256 + threadIdx.x;
    if (i < n) {
        float s = g[i] * rsqrtf(v[i] + 1e-5f);
        sc[i] = s;
        sh[i] = (cb[i] - m[i]) * s + be[i];
    }
}

// =====================================================================
// 6) Build featc [131072, 160] bf16: [pf | seed2 | pts | zero-pad]
// =====================================================================
__global__ __launch_bounds__(FCK)
void build_featc_kernel(const unsigned short* __restrict__ pf,
                        const float* __restrict__ coarse,   // [BP, 64] (48 valid)
                        unsigned short* __restrict__ featc) {
    const int r = blockIdx.x;           // 0..131071
    const int c = threadIdx.x;          // 0..159
    const int bp = r >> 6, k = r & 63;  // k = point within patch
    unsigned short v;
    if (c < DIM) {
        v = pf[(size_t)bp * DIM + c];
    } else if (c == DIM) {              // seed x = tile(a,2)[k%4]
        v = f2bf((k & 1) ? 0.05f : -0.05f);
    } else if (c == DIM + 1) {          // seed y = repeat(a,2)[k%4]
        v = f2bf(((k >> 1) & 1) ? 0.05f : -0.05f);
    } else if (c < DIM + 5) {           // pts = coarse[bp, k/4, axis]
        v = f2bf(coarse[(size_t)bp * W3NP + (k >> 2) * 3 + (c - (DIM + 2))]);
    } else {
        v = 0;                          // K padding 133..159
    }
    featc[(size_t)r * FCK + c] = v;
}

// =====================================================================
// 7) fine = h2 @ cw3 + cb3 + pts  (N=3: VALU dot, one block per row)
// =====================================================================
__global__ __launch_bounds__(128)
void fine_kernel(const unsigned short* __restrict__ h2, const float* __restrict__ cw3,
                 const float* __restrict__ cb3, const float* __restrict__ coarse,
                 float* __restrict__ fine) {
    const int r = blockIdx.x;
    const int t = threadIdx.x;
    float a0 = 0.f, a1 = 0.f, a2 = 0.f;
    for (int j = t; j < DFF; j += 128) {
        float h = bf2f(h2[(size_t)r * DFF + j]);
        a0 += h * cw3[j * 3 + 0];
        a1 += h * cw3[j * 3 + 1];
        a2 += h * cw3[j * 3 + 2];
    }
    __shared__ float red[3][128];
    red[0][t] = a0; red[1][t] = a1; red[2][t] = a2;
    __syncthreads();
    for (int off = 64; off > 0; off >>= 1) {
        if (t < off) {
            red[0][t] += red[0][t + off];
            red[1][t] += red[1][t + off];
            red[2][t] += red[2][t + off];
        }
        __syncthreads();
    }
    if (t == 0) {
        const int bp = r >> 6, k = r & 63;
        const float* pts = &coarse[(size_t)bp * W3NP + (k >> 2) * 3];
        fine[(size_t)r * 3 + 0] = red[0][0] + cb3[0] + pts[0];
        fine[(size_t)r * 3 + 1] = red[1][0] + cb3[1] + pts[1];
        fine[(size_t)r * 3 + 2] = red[2][0] + cb3[2] + pts[2];
    }
}

// =====================================================================
// 8) Chamfer: per-patch 64x64 distances; deterministic tree reductions.
// =====================================================================
__global__ __launch_bounds__(64)
void chamfer_kernel(const float* __restrict__ fine, const float* __restrict__ gxyz,
                    float* __restrict__ partial) {
    const int bp = blockIdx.x;
    const int t = threadIdx.x;   // 0..63
    __shared__ float ff[KNN][3], gg[KNN][3], red[KNN];
    ff[t][0] = fine[((size_t)bp * KNN + t) * 3 + 0];
    ff[t][1] = fine[((size_t)bp * KNN + t) * 3 + 1];
    ff[t][2] = fine[((size_t)bp * KNN + t) * 3 + 2];
    gg[t][0] = gxyz[((size_t)bp * KNN + t) * 3 + 0];
    gg[t][1] = gxyz[((size_t)bp * KNN + t) * 3 + 1];
    gg[t][2] = gxyz[((size_t)bp * KNN + t) * 3 + 2];
    __syncthreads();
    float m1 = 1e30f, m2 = 1e30f;
    const float fx = ff[t][0], fy = ff[t][1], fz = ff[t][2];
    const float gx = gg[t][0], gy = gg[t][1], gz = gg[t][2];
    for (int j = 0; j < KNN; ++j) {
        float dx = fx - gg[j][0], dy = fy - gg[j][1], dz = fz - gg[j][2];
        m1 = fminf(m1, dx * dx + dy * dy + dz * dz);
        dx = gx - ff[j][0]; dy = gy - ff[j][1]; dz = gz - ff[j][2];
        m2 = fminf(m2, dx * dx + dy * dy + dz * dz);
    }
    red[t] = m1 + m2;
    __syncthreads();
    for (int off = 32; off > 0; off >>= 1) {
        if (t < off) red[t] += red[t + off];
        __syncthreads();
    }
    if (t == 0) partial[bp] = red[0];
}

__global__ __launch_bounds__(256)
void final_reduce_kernel(const float* __restrict__ partial, float* __restrict__ out) {
    const int t = threadIdx.x;
    float s = 0.f;
    for (int i = t; i < BP; i += 256) s += partial[i];
    __shared__ float red[256];
    red[t] = s;
    __syncthreads();
    for (int off = 128; off > 0; off >>= 1) {
        if (t < off) red[t] += red[t + off];
        __syncthreads();
    }
    if (t == 0) out[0] = red[0] * (1.0f / (float)RFINE);
}

// =====================================================================
// Host-side orchestration
// =====================================================================
extern "C" void kernel_launch(void* const* d_in, const int* in_sizes, int n_in,
                              void* d_out, int out_size, void* d_ws, size_t ws_size,
                              hipStream_t stream) {
    const float* xyz  = (const float*)d_in[0];
    const float* feat = (const float*)d_in[1];
    const float* w1   = (const float*)d_in[2];
    const float* b1   = (const float*)d_in[3];
    const float* w2   = (const float*)d_in[4];
    const float* b2   = (const float*)d_in[5];
    const float* w3   = (const float*)d_in[6];
    const float* b3   = (const float*)d_in[7];
    const float* cw1  = (const float*)d_in[8];
    const float* cb1  = (const float*)d_in[9];
    const float* g1   = (const float*)d_in[10];
    const float* be1  = (const float*)d_in[11];
    const float* m1   = (const float*)d_in[12];
    const float* v1   = (const float*)d_in[13];
    const float* cw2  = (const float*)d_in[14];
    const float* cb2  = (const float*)d_in[15];
    const float* g2   = (const float*)d_in[16];
    const float* be2  = (const float*)d_in[17];
    const float* m2   = (const float*)d_in[18];
    const float* v2   = (const float*)d_in[19];
    const float* cw3  = (const float*)d_in[20];
    const float* cb3  = (const float*)d_in[21];
    float* out = (float*)d_out;

    // workspace carve-out (deterministic layout, 256B aligned)
    uintptr_t base = (uintptr_t)d_ws;
    size_t off = 0;
    auto alloc = [&](size_t bytes) {
        void* p = (void*)(base + off);
        off = (off + bytes + 255) & ~(size_t)255;
        return p;
    };
    int*            fidx   = (int*)           alloc(sizeof(int) * BP);
    int*            gidx   = (int*)           alloc(sizeof(int) * (size_t)BP * KNN);
    float*          gxyz   = (float*)         alloc(sizeof(float) * (size_t)RFINE * 3);
    unsigned short* pf_b   = (unsigned short*)alloc(2ull * BP * DIM);
    unsigned short* w1t    = (unsigned short*)alloc(2ull * DFF * DIM);    // [1024,128]
    unsigned short* w2t    = (unsigned short*)alloc(2ull * DFF * DFF);    // [1024,1024]
    unsigned short* w3t    = (unsigned short*)alloc(2ull * W3NP * DFF);   // [64,1024]
    unsigned short* cw1t   = (unsigned short*)alloc(2ull * DFF * FCK);    // [1024,160]
    unsigned short* cw2t   = (unsigned short*)alloc(2ull * DFF * DFF);    // [1024,1024]
    float*          sc1    = (float*)alloc(sizeof(float) * DFF);
    float*          sh1    = (float*)alloc(sizeof(float) * DFF);
    float*          sc2    = (float*)alloc(sizeof(float) * DFF);
    float*          sh2    = (float*)alloc(sizeof(float) * DFF);
    float*          sc3    = (float*)alloc(sizeof(float) * W3NP);
    float*          sh3    = (float*)alloc(sizeof(float) * W3NP);
    float*          sc4    = (float*)alloc(sizeof(float) * DFF);
    float*          sh4    = (float*)alloc(sizeof(float) * DFF);
    float*          sc5    = (float*)alloc(sizeof(float) * DFF);
    float*          sh5    = (float*)alloc(sizeof(float) * DFF);
    unsigned short* hA     = (unsigned short*)alloc(2ull * BP * DFF);
    unsigned short* hB     = (unsigned short*)alloc(2ull * BP * DFF);
    float*          coarse = (float*)         alloc(sizeof(float) * (size_t)BP * W3NP);
    unsigned short* featc  = (unsigned short*)alloc(2ull * RFINE * FCK);
    unsigned short* h1b    = (unsigned short*)alloc(2ull * RFINE * DFF);
    unsigned short* h2b    = (unsigned short*)alloc(2ull * RFINE * DFF);
    float*          fine   = (float*)         alloc(sizeof(float) * (size_t)RFINE * 3);
    float*          partial= (float*)         alloc(sizeof(float) * BP);
    (void)ws_size; (void)n_in; (void)in_sizes; (void)out_size;

    // ---- geometry front-end ----
    fps_kernel<<<BATCH, 256, 0, stream>>>(xyz, fidx);
    topk_kernel<<<BP, 256, 0, stream>>>(xyz, fidx, gidx, gxyz);
    gather_max_kernel<<<BP, 128, 0, stream>>>(feat, gidx, pf_b);

    // ---- weight transpose+convert / folding ----
    // dst[No,Ko] <- src[Ks,Ns]^T with zero pad
    cvt_transpose_kernel<<<(DFF * DIM + 255) / 256, 256, 0, stream>>>(
        w1, w1t, DIM, DFF, DIM, DFF);                 // w1 [128,1024] -> [1024,128]
    cvt_transpose_kernel<<<(DFF * DFF + 255) / 256, 256, 0, stream>>>(
        w2, w2t, DFF, DFF, DFF, DFF);                 // w2 [1024,1024] -> [1024,1024]
    cvt_transpose_kernel<<<(W3NP * DFF + 255) / 256, 256, 0, stream>>>(
        w3, w3t, DFF, W3NP, DFF, 48);                 // w3 [1024,48] -> [64,1024] pad N
    cvt_transpose_kernel<<<(DFF * FCK + 255) / 256, 256, 0, stream>>>(
        cw1, cw1t, FCK, DFF, 133, DFF);               // cw1 [133,1024] -> [1024,160] pad K
    cvt_transpose_kernel<<<(DFF * DFF + 255) / 256, 256, 0, stream>>>(
        cw2, cw2t, DFF, DFF, DFF, DFF);               // cw2 [1024,1024] -> [1024,1024]
    fold_bias_kernel<<<(DFF + 255) / 256, 256, 0, stream>>>(b1, sc1, sh1, DFF, DFF);
    fold_bias_kernel<<<(DFF + 255) / 256, 256, 0, stream>>>(b2, sc2, sh2, DFF, DFF);
    fold_bias_kernel<<<1, 256, 0, stream>>>(b3, sc3, sh3, W3NP, 48);
    fold_bn_kernel<<<(DFF + 255) / 256, 256, 0, stream>>>(cb1, g1, be1, m1, v1, sc4, sh4, DFF);
    fold_bn_kernel<<<(DFF + 255) / 256, 256, 0, stream>>>(cb2, g2, be2, m2, v2, sc5, sh5, DFF);

    // ---- FoldingNet MLP (WMMA) ----
    // h = relu(pf @ w1 + b1)       M=2048 N=1024 K=128
    gemm_bf16_wmma<<<dim3(DFF / GBN, BP / GBM), 256, 0, stream>>>(
        pf_b, w1t, sc1, sh1, nullptr, hA, BP, DFF, DIM, 1);
    // h = relu(h @ w2 + b2)        M=2048 N=1024 K=1024
    gemm_bf16_wmma<<<dim3(DFF / GBN, BP / GBM), 256, 0, stream>>>(
        hA, w2t, sc2, sh2, nullptr, hB, BP, DFF, DFF, 1);
    // coarse = h @ w3 + b3         M=2048 N=64(pad) K=1024, f32 out
    gemm_bf16_wmma<<<dim3(W3NP / GBN, BP / GBM), 256, 0, stream>>>(
        hB, w3t, sc3, sh3, coarse, nullptr, BP, W3NP, DFF, 0);

    // ---- final_conv (WMMA, dominant cost) ----
    build_featc_kernel<<<RFINE, FCK, 0, stream>>>(pf_b, coarse, featc);
    // h1 = relu(bn1(featc @ cw1 + cb1))   M=131072 N=1024 K=160
    gemm_bf16_wmma<<<dim3(DFF / GBN, RFINE / GBM), 256, 0, stream>>>(
        featc, cw1t, sc4, sh4, nullptr, h1b, RFINE, DFF, FCK, 1);
    // h2 = relu(bn2(h1 @ cw2 + cb2))      M=131072 N=1024 K=1024  <- 275 GFLOP
    gemm_bf16_wmma<<<dim3(DFF / GBN, RFINE / GBM), 256, 0, stream>>>(
        h1b, cw2t, sc5, sh5, nullptr, h2b, RFINE, DFF, DFF, 1);

    // ---- fine points + Chamfer loss ----
    fine_kernel<<<RFINE, 128, 0, stream>>>(h2b, cw3, cb3, coarse, fine);
    chamfer_kernel<<<BP, 64, 0, stream>>>(fine, gxyz, partial);
    final_reduce_kernel<<<1, 256, 0, stream>>>(partial, out);
}
